// GraphAttentionLayer_31688268710276
// MI455X (gfx1250) — matrix operand order repaired and checked by reading, not compiled, not run on hardware
//
#include <hip/hip_runtime.h>
#include <math.h>

typedef __attribute__((ext_vector_type(16))) _Float16 v16h;
typedef __attribute__((ext_vector_type(8)))  _Float16 v8h;
typedef __attribute__((ext_vector_type(8)))  float    v8f;

#define BQ   8
#define NQ   2048
#define FQ   256
#define BNQ  (BQ * NQ)     // 16384 rows total
#define CHQ  8             // i-chunks for column-softmax partials

// ---------------------------------------------------------------------------
// Kernel 0: W (256x256 f32, [k][o]) -> Wt f16 transposed ([o][k])
// ---------------------------------------------------------------------------
__global__ void k_convert_w(const float* __restrict__ W, _Float16* __restrict__ Wt) {
    int i = blockIdx.x * blockDim.x + threadIdx.x;
    if (i < FQ * FQ) {
        int k = i >> 8, o = i & (FQ - 1);
        Wt[o * FQ + k] = (_Float16)W[i];      // coalesced read, scattered 64K store (one-off)
    }
}

// ---------------------------------------------------------------------------
// Kernel 1: h = inp @ W via v_wmma_f32_16x16x32_f16; result stored TRANSPOSED
// as hT[b][o][n] f16. Block = 512 threads (16 waves) owns 16 rows x 256 cols.
// B-operand fragments are contiguous 32B loads from Wt; the C layout
// (col per lane, 8 consecutive rows per lane-half) makes the transposed
// store a single contiguous b128 per lane.
// ---------------------------------------------------------------------------
__global__ void k_gemm_h(const float* __restrict__ inp,
                         const _Float16* __restrict__ Wt,
                         _Float16* __restrict__ hT) {
    __shared__ _Float16 As[16][FQ + 8];

    const int bn0 = blockIdx.x * 16;
    const int b   = bn0 / NQ;
    const int n0  = bn0 % NQ;
    const int tid = threadIdx.x;

    // stage 16x256 A slab (f32 -> f16) with float4 loads (slab is contiguous)
    for (int e = tid; e < 16 * FQ / 4; e += 512) {
        int r = e >> 6, k4 = e & 63;
        float4 v = ((const float4*)inp)[(size_t)(bn0 + r) * (FQ / 4) + k4];
        As[r][k4 * 4 + 0] = (_Float16)v.x;
        As[r][k4 * 4 + 1] = (_Float16)v.y;
        As[r][k4 * 4 + 2] = (_Float16)v.z;
        As[r][k4 * 4 + 3] = (_Float16)v.w;
    }
    __syncthreads();

    const int w    = tid >> 5;
    const int lane = tid & 31;
    const int g    = lane >> 4;
    const int arow = lane & 15;
    const int col  = (w << 4) + (lane & 15);

    v8f acc = {};
#pragma unroll
    for (int k0 = 0; k0 < FQ; k0 += 32) {
        v16h af;
#pragma unroll
        for (int i = 0; i < 16; ++i) {
            int k = k0 + (g << 4) + i;
            af[i] = As[arow][k];
        }
        // B fragment: lane needs K = k0+16g .. +16 at fixed col -> contiguous in Wt
        v16h bf = *(const v16h*)(&Wt[(size_t)col * FQ + k0 + (g << 4)]);
        acc = __builtin_amdgcn_wmma_f32_16x16x32_f16(
            false, af, false, bf, (short)0, acc, false, false);
    }

    // transposed store: rows r+8g are 8 consecutive n -> one b128 per lane
    v8h hv;
#pragma unroll
    for (int r = 0; r < 8; ++r) hv[r] = (_Float16)acc[r];
    *(v8h*)(&hT[((size_t)b * FQ + col) * NQ + n0 + (g << 3)]) = hv;
}

// ---------------------------------------------------------------------------
// Kernel 2: s1[row]=h[row,:].a1, s2[row]=h[row,:].a2 from hT.
// Thread per row-index n: for each o all threads read consecutive n -> coalesced.
// ---------------------------------------------------------------------------
__global__ void k_scores(const _Float16* __restrict__ hT,
                         const float* __restrict__ a,
                         float* __restrict__ s1, float* __restrict__ s2) {
    const int gid = blockIdx.x * blockDim.x + threadIdx.x;
    if (gid >= BNQ) return;
    const int b = gid >> 11;          // /NQ
    const int n = gid & (NQ - 1);
    const _Float16* hb = hT + (size_t)b * FQ * NQ + n;

    float d1 = 0.f, d2 = 0.f;
#pragma unroll 4
    for (int o = 0; o < FQ; ++o) {
        float v = (float)hb[(size_t)o * NQ];
        d1 += v * a[o];
        d2 += v * a[FQ + o];
    }
    s1[gid] = d1;
    s2[gid] = d2;
}

// ---------------------------------------------------------------------------
// Kernel 3a: partial online-softmax column stats over an i-chunk.
// ---------------------------------------------------------------------------
__global__ void k_colstats_partial(const int* __restrict__ adj,
                                   const float* __restrict__ s1,
                                   const float* __restrict__ s2,
                                   float* __restrict__ pm, float* __restrict__ pz) {
    const int bj    = blockIdx.x % (BQ * (NQ / 256));
    const int chunk = blockIdx.x / (BQ * (NQ / 256));
    const int b     = bj / (NQ / 256);
    const int j     = (bj % (NQ / 256)) * 256 + threadIdx.x;

    const float s2j = s2[b * NQ + j];
    const int* ac = adj + (size_t)b * NQ * NQ + j;
    const float* s1b = s1 + b * NQ;

    float m = -3.0e38f, Z = 0.f;
    const int i0 = chunk * (NQ / CHQ);
    for (int i = i0; i < i0 + NQ / CHQ; ++i) {
        float v;
        if (ac[(size_t)i * NQ] > 0) {
            float t = s1b[i] + s2j;
            v = (t >= 0.f) ? t : 0.4f * t;
        } else {
            v = -1.0e20f;
        }
        if (v > m) { Z *= __expf(m - v); m = v; }
        Z += __expf(v - m);
    }
    pm[(size_t)chunk * BNQ + b * NQ + j] = m;
    pz[(size_t)chunk * BNQ + b * NQ + j] = Z;
}

// ---------------------------------------------------------------------------
// Kernel 3b: combine partials -> colmax, 1/colsum
// ---------------------------------------------------------------------------
__global__ void k_colstats_combine(const float* __restrict__ pm,
                                   const float* __restrict__ pz,
                                   float* __restrict__ cmax, float* __restrict__ crsum) {
    const int idx = blockIdx.x * blockDim.x + threadIdx.x;
    if (idx >= BNQ) return;
    float m = -3.0e38f;
#pragma unroll
    for (int c = 0; c < CHQ; ++c) m = fmaxf(m, pm[(size_t)c * BNQ + idx]);
    float Z = 0.f;
#pragma unroll
    for (int c = 0; c < CHQ; ++c) Z += pz[(size_t)c * BNQ + idx] * __expf(pm[(size_t)c * BNQ + idx] - m);
    cmax[idx]  = m;
    crsum[idx] = 1.0f / fmaxf(Z, 1e-30f);
}

// ---------------------------------------------------------------------------
// Kernel 4: h' = att @ h fused: async double-buffered adj tiles -> LDS,
// P tile (16x32 f16) built in LDS, WMMA vs contiguous hT fragments,
// GELU epilogue via LDS-staged coalesced float4 stores.
// ---------------------------------------------------------------------------
__global__ void k_aggregate_gelu(const int* __restrict__ adj,
                                 const _Float16* __restrict__ hT,
                                 const float* __restrict__ s1,
                                 const float* __restrict__ s2,
                                 const float* __restrict__ cmax,
                                 const float* __restrict__ crsum,
                                 float* __restrict__ out) {
    __shared__ _Float16 pt[16][40];          // 16x32 P tile (padded)
    __shared__ int      adjLds[2][16][32];   // double-buffered adj tile
    __shared__ float    otile[16][FQ + 8];   // output staging

    const int b   = blockIdx.x / (NQ / 16);
    const int i0  = (blockIdx.x % (NQ / 16)) * 16;
    const int tid = threadIdx.x;

    const int pr = tid >> 5;   // row within tile
    const int pj = tid & 31;   // j within tile

    const int lane = tid & 31;
    const int g    = lane >> 4;
    const int arow = lane & 15;
    const int col  = ((tid >> 5) << 4) + (lane & 15);

    const int*   adjb = adj + (size_t)b * NQ * NQ;
    const float* s2b  = s2 + b * NQ;
    const float* mb   = cmax + b * NQ;
    const float* rb   = crsum + b * NQ;
    const _Float16* hcol = hT + ((size_t)b * FQ + col) * NQ;  // fixed column of h

    const float s1i = s1[b * NQ + i0 + pr];
    const float s1i_lr = 0.4f;   // negative slope

    // async-copy coordinates: 128 threads move 16 rows x 32 ints (16B chunks)
    const int crow = tid >> 3;         // 0..15
    const int cc4  = (tid & 7) << 2;   // 0,4,...,28

    // prologue: issue async load of adj tile j0=0 into buffer 0
    if (tid < 128) {
        const int* src = &adjb[(size_t)(i0 + crow) * NQ + cc4];
        unsigned int dst = (unsigned int)(size_t)&adjLds[0][crow][cc4];
        asm volatile("global_load_async_to_lds_b128 %0, %1, off"
                     :: "v"(dst), "v"(src) : "memory");
    }

    v8f acc = {};
    for (int it = 0; it < NQ / 32; ++it) {
        const int j0  = it * 32;
        const int cur = it & 1;

        if (tid < 128) {
            if (it < NQ / 32 - 1) {
                // issue next tile into the other buffer, then retire current
                const int* src = &adjb[(size_t)(i0 + crow) * NQ + j0 + 32 + cc4];
                unsigned int dst = (unsigned int)(size_t)&adjLds[cur ^ 1][crow][cc4];
                asm volatile("global_load_async_to_lds_b128 %0, %1, off"
                             :: "v"(dst), "v"(src) : "memory");
                asm volatile("s_wait_asynccnt 0x1" ::: "memory");
            } else {
                asm volatile("s_wait_asynccnt 0x0" ::: "memory");
            }
        }
        __syncthreads();   // adj buffer visible; previous pt reads complete

        // build P tile from LDS adj
        {
            int j = j0 + pj;
            int av = adjLds[cur][pr][pj];
            float p = 0.f;
            if (av > 0) {
                float t = s1i + s2b[j];
                t = (t >= 0.f) ? t : s1i_lr * t;
                p = __expf(t - mb[j]) * rb[j];
            }
            pt[pr][pj] = (_Float16)p;
        }
        __syncthreads();   // pt visible

        // A fragment from LDS P tile (one contiguous 16-halfword run per lane)
        v16h af;
#pragma unroll
        for (int i = 0; i < 16; ++i) {
            int k = (g << 4) + i;
            af[i] = pt[arow][k];
        }
        // B fragment: contiguous 32B run of hT at fixed column
        v16h bf = *(const v16h*)(&hcol[j0 + (g << 4)]);
        if (it < NQ / 32 - 1)
            __builtin_prefetch(&hcol[j0 + 32 + (g << 4)], 0, 1);

        acc = __builtin_amdgcn_wmma_f32_16x16x32_f16(
            false, af, false, bf, (short)0, acc, false, false);
    }

    // stage result tile in LDS, then GELU + coalesced float4 stores
#pragma unroll
    for (int r = 0; r < 8; ++r) otile[r + (g << 3)][col] = acc[r];
    __syncthreads();

    float* outb = out + ((size_t)b * NQ + i0) * FQ;   // contiguous 16x256 slab
    for (int e = tid; e < 16 * FQ / 4; e += 512) {
        int r = e >> 6, c4 = (e & 63) << 2;
        float4 x = *(const float4*)&otile[r][c4];
        float4 y;
        y.x = 0.5f * x.x * (1.0f + erff(x.x * 0.70710678118654752f));
        y.y = 0.5f * x.y * (1.0f + erff(x.y * 0.70710678118654752f));
        y.z = 0.5f * x.z * (1.0f + erff(x.z * 0.70710678118654752f));
        y.w = 0.5f * x.w * (1.0f + erff(x.w * 0.70710678118654752f));
        *(float4*)&outb[(size_t)r * FQ + c4] = y;
    }
}

// ---------------------------------------------------------------------------
// Launch
// ---------------------------------------------------------------------------
extern "C" void kernel_launch(void* const* d_in, const int* in_sizes, int n_in,
                              void* d_out, int out_size, void* d_ws, size_t ws_size,
                              hipStream_t stream) {
    const float* inp = (const float*)d_in[0];   // (8,2048,256) f32
    const int*   adj = (const int*)d_in[1];     // (8,2048,2048) i32
    const float* W   = (const float*)d_in[2];   // (256,256) f32
    const float* a   = (const float*)d_in[3];   // (512,1) f32
    float* out = (float*)d_out;

    char* ws = (char*)d_ws;
    size_t off = 0;
    _Float16* Wt   = (_Float16*)(ws + off); off += (size_t)FQ * FQ * 2;        // 128 KB
    _Float16* hT   = (_Float16*)(ws + off); off += (size_t)BNQ * FQ * 2;       // 8 MB
    float*    s1   = (float*)(ws + off);    off += (size_t)BNQ * 4;
    float*    s2   = (float*)(ws + off);    off += (size_t)BNQ * 4;
    float*    cmax = (float*)(ws + off);    off += (size_t)BNQ * 4;
    float*    crs  = (float*)(ws + off);    off += (size_t)BNQ * 4;
    float*    pm   = (float*)(ws + off);    off += (size_t)CHQ * BNQ * 4;
    float*    pz   = (float*)(ws + off);    off += (size_t)CHQ * BNQ * 4;

    k_convert_w<<<(FQ * FQ + 255) / 256, 256, 0, stream>>>(W, Wt);

    k_gemm_h<<<BNQ / 16, 512, 0, stream>>>(inp, Wt, hT);

    k_scores<<<BNQ / 256, 256, 0, stream>>>(hT, a, s1, s2);

    k_colstats_partial<<<CHQ * BQ * (NQ / 256), 256, 0, stream>>>(adj, s1, s2, pm, pz);

    k_colstats_combine<<<(BNQ + 255) / 256, 256, 0, stream>>>(pm, pz, cmax, crs);

    k_aggregate_gelu<<<BQ * (NQ / 16), 512, 0, stream>>>(adj, hT, s1, s2, cmax, crs, out);
}
